// Conv2dModulated_28836410426123
// MI455X (gfx1250) — matrix operand order repaired and checked by reading, not compiled, not run on hardware
//
#include <hip/hip_runtime.h>
#include <hip/hip_bf16.h>

// ---------------------------------------------------------------------------
// Problem constants (from reference)
// ---------------------------------------------------------------------------
#define BN   16
#define CIN  512
#define COUT 512
#define LAT  512
#define HW   32
#define NPIX (HW * HW)   // 1024
#define HO   65          // conv-transpose output
#define WO   65
#define HF   64          // final output
#define WF   64
#define ZSLOT (BN * NPIX)   // index of the zero pixel row appended to xsT

typedef __bf16 bf16_t;
typedef __attribute__((ext_vector_type(16))) __bf16 v16bf;
typedef __attribute__((ext_vector_type(8)))  float  v8f;

union Frag32B {
    uint4 q[2];
    v16bf v;
};

// ---------------------------------------------------------------------------
// K1: style s[b,ci] = w[b,:] . affine_w[ci,:] + affine_b[ci] + 1
// ---------------------------------------------------------------------------
__global__ void style_kernel(const float* __restrict__ w,
                             const float* __restrict__ aw,
                             const float* __restrict__ ab,
                             float* __restrict__ s) {
    int idx = blockIdx.x * blockDim.x + threadIdx.x;
    if (idx >= BN * CIN) return;
    int b = idx / CIN, ci = idx % CIN;
    const float* wr = w + b * LAT;
    const float* ar = aw + (long)ci * LAT;
    float acc = 0.f;
#pragma unroll 8
    for (int l = 0; l < LAT; ++l) acc += wr[l] * ar[l];
    s[idx] = acc + ab[ci] + 1.0f;
}

// ---------------------------------------------------------------------------
// K2: demod d[b,co] = rsqrt( sum_ci s[b,ci]^2 * sum_k weight[co,ci,k]^2 + eps )
// ---------------------------------------------------------------------------
__global__ void demod_kernel(const float* __restrict__ s,
                             const float* __restrict__ weight,
                             float* __restrict__ dsc) {
    int idx = blockIdx.x * blockDim.x + threadIdx.x;
    if (idx >= BN * COUT) return;
    int b = idx / COUT, co = idx % COUT;
    const float* sr = s + b * CIN;
    const float* wr = weight + (long)co * CIN * 9;
    float acc = 0.f;
    for (int ci = 0; ci < CIN; ++ci) {
        float w2 = 0.f;
#pragma unroll
        for (int k = 0; k < 9; ++k) {
            float v = wr[ci * 9 + k];
            w2 += v * v;
        }
        float sv = sr[ci];
        acc += sv * sv * w2;
    }
    dsc[idx] = rsqrtf(acc + 1e-8f);
}

// ---------------------------------------------------------------------------
// K3: modulated input, transposed to [b][pix][ci] bf16 (ci contiguous).
//     One extra zeroed pixel row at slot ZSLOT (boundary zero page).
// ---------------------------------------------------------------------------
__global__ void modx_kernel(const float* __restrict__ x,
                            const float* __restrict__ s,
                            bf16_t* __restrict__ xsT) {
    long idx = (long)blockIdx.x * blockDim.x + threadIdx.x;
    if (idx >= (long)(ZSLOT + 1) * CIN) return;
    if (idx >= (long)ZSLOT * CIN) {       // zero page
        xsT[idx] = (bf16_t)0.0f;
        return;
    }
    int ci = (int)(idx % CIN);
    long t = idx / CIN;
    int pix = (int)(t % NPIX);
    int b = (int)(t / NPIX);
    float v = x[((long)b * CIN + ci) * NPIX + pix] * s[b * CIN + ci];
    xsT[idx] = (bf16_t)v;
}

// ---------------------------------------------------------------------------
// K4: weights repacked to [kk(9)][co][ci] bf16 (ci contiguous)
// ---------------------------------------------------------------------------
__global__ void wprep_kernel(const float* __restrict__ weight,
                             bf16_t* __restrict__ wT) {
    long idx = (long)blockIdx.x * blockDim.x + threadIdx.x;
    if (idx >= (long)9 * COUT * CIN) return;
    int ci = (int)(idx % CIN);
    long t = idx / CIN;
    int co = (int)(t % COUT);
    int kk = (int)(t / COUT);
    wT[idx] = (bf16_t)weight[((long)co * CIN + ci) * 9 + kk];
}

// ---------------------------------------------------------------------------
// K5: main WMMA implicit GEMM per output-parity class (templated).
//   Wave tile: 32 pix (2 M-subtiles) x 64 co (4 N-subtiles) = 8 v8f accums.
//   Block 256 = 8 waves: 4 pix-groups x 2 co-groups -> 128 pix x 128 co.
//   Invalid input pixels resolve to the xsT zero page (no EXEC predication).
// ---------------------------------------------------------------------------
template <int PARY, int PARX>
__global__ __launch_bounds__(256) void wmma_conv_kernel(
        const bf16_t* __restrict__ xsT, const bf16_t* __restrict__ wT,
        const float* __restrict__ dsc, float* __restrict__ y65) {
    constexpr int nY = PARY ? 32 : 33;
    constexpr int nX = PARX ? 32 : 33;
    constexpr int npix = nY * nX;
    constexpr int ntY = PARY ? 1 : 2;
    constexpr int ntX = PARX ? 1 : 2;

    const int wave = threadIdx.x >> 5;
    const int lane = threadIdx.x & 31;
    const int lhalf = lane >> 4;   // 0 or 1 (half-wave)
    const int lmod = lane & 15;

    const int b = blockIdx.z;
    const int mbase = blockIdx.x * 128 + (wave & 3) * 32;   // 32 pix per wave
    const int cobase = blockIdx.y * 128 + (wave >> 2) * 64; // 64 co per wave

    v8f acc[2][4];
#pragma unroll
    for (int mi = 0; mi < 2; ++mi)
#pragma unroll
        for (int ni = 0; ni < 4; ++ni)
            acc[mi][ni] = (v8f){0.f, 0.f, 0.f, 0.f, 0.f, 0.f, 0.f, 0.f};

#pragma unroll
    for (int ty = 0; ty < ntY; ++ty) {
        const int kky = PARY ? 1 : (ty == 0 ? 0 : 2);
#pragma unroll
        for (int tx = 0; tx < ntX; ++tx) {
            const int kkx = PARX ? 1 : (tx == 0 ? 0 : 2);
            const int kkIdx = kky * 3 + kkx;

            // Per-M-subtile A row pointers (zero page when OOB).
            const bf16_t* aptr[2];
#pragma unroll
            for (int mi = 0; mi < 2; ++mi) {
                const int m = mbase + mi * 16 + lmod;   // A row = lane % 16
                const int py = m / nX;
                const int px = m % nX;
                const int iy = py - ty;
                const int ix = px - tx;
                const bool valid = (m < npix) && (iy >= 0) && (iy < HW) &&
                                   (ix >= 0) && (ix < HW);
                const long slot =
                    valid ? ((long)b * NPIX + iy * HW + ix) : (long)ZSLOT;
                aptr[mi] = xsT + slot * CIN;
            }
            // Per-N-subtile B row pointers (col = lane % 16).
            const bf16_t* bptr[4];
#pragma unroll
            for (int ni = 0; ni < 4; ++ni)
                bptr[ni] =
                    wT + (((long)kkIdx * COUT + cobase + ni * 16 + lmod) * CIN);

            for (int ci0 = 0; ci0 < CIN; ci0 += 32) {
                // A fragments (16-bit A 16x32 layout):
                //   halves 0..7  -> K = ci0 +      8*lhalf + j
                //   halves 8..15 -> K = ci0 + 16 + 8*lhalf + (j-8)
                Frag32B af[2];
#pragma unroll
                for (int mi = 0; mi < 2; ++mi) {
                    af[mi].q[0] = *(const uint4*)(aptr[mi] + ci0 + 8 * lhalf);
                    af[mi].q[1] =
                        *(const uint4*)(aptr[mi] + ci0 + 16 + 8 * lhalf);
                }
                // B fragments (16-bit B 32x16): same K order as A fragments
                Frag32B bf[4];
#pragma unroll
                for (int ni = 0; ni < 4; ++ni) {
                    bf[ni].q[0] = *(const uint4*)(bptr[ni] + ci0 + 8 * lhalf);
                    bf[ni].q[1] =
                        *(const uint4*)(bptr[ni] + ci0 + 16 + 8 * lhalf);
                }
                // 8 independent WMMAs (no RAW chains -> XDL pipelining).
#pragma unroll
                for (int mi = 0; mi < 2; ++mi)
#pragma unroll
                    for (int ni = 0; ni < 4; ++ni)
                        acc[mi][ni] = __builtin_amdgcn_wmma_f32_16x16x32_bf16(
                            false, af[mi].v, false, bf[ni].v, (short)0,
                            acc[mi][ni], false, false);
            }
        }
    }

    // D layout: col N = lane%16 (= co), row M = v + 8*(lane/16)
#pragma unroll
    for (int ni = 0; ni < 4; ++ni) {
        const int co = cobase + ni * 16 + lmod;
        const float dv = dsc[b * COUT + co];
#pragma unroll
        for (int mi = 0; mi < 2; ++mi) {
#pragma unroll
            for (int v = 0; v < 8; ++v) {
                const int mm = mbase + mi * 16 + v + 8 * lhalf;
                if (mm < npix) {
                    const int py2 = mm / nX, px2 = mm % nX;
                    const int oy = 2 * py2 + PARY, ox = 2 * px2 + PARX;
                    y65[(((long)b * COUT + co) * HO + oy) * WO + ox] =
                        dv * acc[mi][ni][v];
                }
            }
        }
    }
}

// ---------------------------------------------------------------------------
// K6: 4x4 depthwise blur ([1,3,3,1] x [1,3,3,1] / 16, symmetric) + bias
// ---------------------------------------------------------------------------
__global__ void blur_kernel(const float* __restrict__ y65,
                            const float* __restrict__ bias,
                            float* __restrict__ out) {
    long idx = (long)blockIdx.x * blockDim.x + threadIdx.x;
    if (idx >= (long)BN * COUT * HF * WF) return;
    const int X = (int)(idx & (WF - 1));
    long t = idx >> 6;
    const int Y = (int)(t & (HF - 1));
    t >>= 6;
    const int co = (int)(t % COUT);
    const int b = (int)(t / COUT);

    const float tap[4] = {1.f, 3.f, 3.f, 1.f};
    const float* src = y65 + ((long)(b * COUT + co)) * HO * WO;
    float acc = 0.f;
#pragma unroll
    for (int u = 0; u < 4; ++u) {
        const int yy = Y - 1 + u;
        if (yy < 0 || yy >= HO) continue;
        float rowacc = 0.f;
#pragma unroll
        for (int v = 0; v < 4; ++v) {
            const int xx = X - 1 + v;
            if (xx < 0 || xx >= WO) continue;
            rowacc += tap[v] * src[yy * WO + xx];
        }
        acc += tap[u] * rowacc;
    }
    out[idx] = acc * (1.0f / 16.0f) + bias[co];
}

// ---------------------------------------------------------------------------
// launcher
// ---------------------------------------------------------------------------
extern "C" void kernel_launch(void* const* d_in, const int* in_sizes, int n_in,
                              void* d_out, int out_size, void* d_ws,
                              size_t ws_size, hipStream_t stream) {
    (void)in_sizes; (void)n_in; (void)out_size; (void)ws_size;
    const float* x   = (const float*)d_in[0];
    const float* w   = (const float*)d_in[1];
    const float* wt  = (const float*)d_in[2];   // weight (COUT,CIN,3,3)
    const float* bia = (const float*)d_in[3];
    const float* aw  = (const float*)d_in[4];
    const float* ab  = (const float*)d_in[5];
    float* out = (float*)d_out;

    char* ws = (char*)d_ws;
    const size_t off_s   = 0;                                   // 32 KB
    const size_t off_d   = off_s + (size_t)BN * CIN * 4;        // 32 KB
    const size_t off_wT  = off_d + (size_t)BN * COUT * 4;       // 4.7 MB
    const size_t off_xsT = off_wT + (size_t)9 * COUT * CIN * 2; // 16.8 MB+1KB
    const size_t off_y65 =
        off_xsT + (size_t)(ZSLOT + 1) * CIN * 2;                // 138 MB
    float*  s_buf  = (float*)(ws + off_s);
    float*  d_buf  = (float*)(ws + off_d);
    bf16_t* wT     = (bf16_t*)(ws + off_wT);
    bf16_t* xsT    = (bf16_t*)(ws + off_xsT);
    float*  y65    = (float*)(ws + off_y65);

    // K1 style
    style_kernel<<<(BN * CIN + 255) / 256, 256, 0, stream>>>(w, aw, ab, s_buf);
    // K2 demod scale (needs s)
    demod_kernel<<<(BN * COUT + 255) / 256, 256, 0, stream>>>(s_buf, wt, d_buf);
    // K3 modulated input transpose -> bf16 (+ zero page)
    {
        long n = (long)(ZSLOT + 1) * CIN;
        modx_kernel<<<(unsigned)((n + 255) / 256), 256, 0, stream>>>(x, s_buf,
                                                                     xsT);
    }
    // K4 weight repack -> bf16
    {
        long n = (long)9 * COUT * CIN;
        wprep_kernel<<<(unsigned)((n + 255) / 256), 256, 0, stream>>>(wt, wT);
    }
    // K5 four parity classes of the stride-2 transpose conv (partition 65x65)
    {
        dim3 blk(256);
        // EE: 33x33 = 1089 pix
        wmma_conv_kernel<0, 0><<<dim3((1089 + 127) / 128, COUT / 128, BN), blk,
                                 0, stream>>>(xsT, wT, d_buf, y65);
        // EO: 33x32 = 1056 pix
        wmma_conv_kernel<0, 1><<<dim3((1056 + 127) / 128, COUT / 128, BN), blk,
                                 0, stream>>>(xsT, wT, d_buf, y65);
        // OE: 32x33 = 1056 pix
        wmma_conv_kernel<1, 0><<<dim3((1056 + 127) / 128, COUT / 128, BN), blk,
                                 0, stream>>>(xsT, wT, d_buf, y65);
        // OO: 32x32 = 1024 pix
        wmma_conv_kernel<1, 1><<<dim3((1024 + 127) / 128, COUT / 128, BN), blk,
                                 0, stream>>>(xsT, wT, d_buf, y65);
    }
    // K6 blur + bias
    {
        long n = (long)BN * COUT * HF * WF;
        blur_kernel<<<(unsigned)((n + 255) / 256), 256, 0, stream>>>(y65, bia,
                                                                     out);
    }
}
